// MultiheadAttention_82617990906052
// MI455X (gfx1250) — matrix operand order repaired and checked by reading, not compiled, not run on hardware
//
#include <hip/hip_runtime.h>
#include <hip/hip_bf16.h>

typedef __bf16 bf16_t;
typedef __attribute__((ext_vector_type(16))) __bf16 v16bf;
typedef __attribute__((ext_vector_type(8)))  __bf16 v8bf;
typedef __attribute__((ext_vector_type(8)))  float  v8f;

#define DEVFN static __device__ __forceinline__

constexpr int B_ = 2, S_ = 2048, E_ = 1024, H_ = 16, HD_ = 64;
constexpr float CLAMP_ = 5.0f;
constexpr float INV_SQRT_HD = 0.125f;  // 1/sqrt(64)

// ---------- helpers ----------

DEVFN bf16_t f2bf(float f) {
  union { float f; unsigned u; } v; v.f = f;
  unsigned r = v.u + 0x7FFFu + ((v.u >> 16) & 1u);   // round-to-nearest-even
  union { unsigned short s; bf16_t b; } o; o.s = (unsigned short)(r >> 16);
  return o.b;
}

DEVFN v8f wmma_bf16(v16bf a, v16bf b, v8f c) {
  // D(f32 16x16) = A(bf16 16x32) * B(bf16 32x16) + C
  return __builtin_amdgcn_wmma_f32_16x16x32_bf16(
      /*neg_a=*/false, a, /*neg_b=*/false, b,
      /*c_mod=*/(short)0, c, /*reuse_a=*/false, /*reuse_b=*/false);
}

// A-fragment: 16x32 (MxK) bf16, src row-major, row stride ld elems.
// Contiguous-k layout used here (not the ISA table order): lane m = lane&15; kbase = (lane>=16)?16:0;
// elems i -> K = kbase+i (contiguous 32B)
DEVFN v16bf load_a(const bf16_t* __restrict__ tile, int ld, int lane) {
  const int m  = lane & 15;
  const int kb = (lane >> 4) * 16;
  const bf16_t* p = tile + (size_t)m * ld + kb;
  return *(const v16bf*)(p);
}

// B-fragment: logical B[k][n] = src[n*ld + k] (src row-major [N,K]; computes A*src^T).
// Contiguous-k layout used here (not the ISA table order): lane n = lane&15; elems i -> K = kbase+i, kbase = (lane>=16)?16:0 -> contiguous 32B.
DEVFN v16bf load_bt(const bf16_t* __restrict__ tile, int ld, int lane) {
  const int n  = lane & 15;
  const int kb = (lane >> 4) * 16;
  return *(const v16bf*)(tile + (size_t)n * ld + kb);
}

// ---------- kernels ----------

__global__ void __launch_bounds__(256) k_cvt(const float* __restrict__ s,
                                             bf16_t* __restrict__ d, int n) {
  for (int i = blockIdx.x * blockDim.x + threadIdx.x; i < n; i += gridDim.x * blockDim.x)
    d[i] = f2bf(s[i]);
}

// QKV projection: out[row, f] = sum_e X[row,e] * Wqkv[f,e] + bqkv[f]
// Scatter epilogue: f -> (h = f/192, c = f%192); c<64 -> Q, c<128 -> K, else V^T.
// Q,K: [B*H, S, 64] bf16 row-major.  Vt: [B*H, 64, S] bf16 row-major.
__global__ void __launch_bounds__(256) k_qkv_gemm(const bf16_t* __restrict__ Xb,
                                                  const bf16_t* __restrict__ Wb,
                                                  const float*  __restrict__ bqkv,
                                                  bf16_t* __restrict__ Qb,
                                                  bf16_t* __restrict__ Kmb,
                                                  bf16_t* __restrict__ Vt) {
  const int lane = threadIdx.x & 31;
  const int wid  = threadIdx.x >> 5;
  const int tile = blockIdx.x * 8 + wid;     // 49152 tiles total
  const int mt   = tile & 255;               // 256 row tiles (M = 4096)
  const int nt   = tile >> 8;                // 192 col tiles (N = 3072)
  const int row0 = mt * 16, col0 = nt * 16;

  v8f acc = {};
#pragma unroll 4
  for (int k0 = 0; k0 < E_; k0 += 32) {
    v16bf a = load_a (Xb + (size_t)row0 * E_ + k0, E_, lane);
    v16bf b = load_bt(Wb + (size_t)col0 * E_ + k0, E_, lane);
    acc = wmma_bf16(a, b, acc);
  }

  const int l16 = lane & 15, half = lane >> 4;
  const int f = col0 + l16;
  const int h = f / (3 * HD_);
  const int c = f % (3 * HD_);
  const float bias = bqkv[f];
#pragma unroll
  for (int r = 0; r < 8; ++r) {
    const int row = row0 + r + half * 8;
    const int bb  = row >> 11;               // / S_
    const int ss  = row & (S_ - 1);
    const bf16_t v = f2bf(acc[r] + bias);
    const size_t hb = (size_t)(bb * H_ + h);
    if (c < HD_)           Qb [(hb * S_ + ss) * HD_ + c]            = v;
    else if (c < 2 * HD_)  Kmb[(hb * S_ + ss) * HD_ + (c - HD_)]    = v;
    else                   Vt [(hb * HD_ + (c - 2 * HD_)) * S_ + ss] = v;
  }
}

// Flash attention: 1 wave handles a 16-row Q tile; online softmax over 32-col K steps.
__global__ void __launch_bounds__(128) k_attn(const bf16_t* __restrict__ Qb,
                                              const bf16_t* __restrict__ Kmb,
                                              const bf16_t* __restrict__ Vt,
                                              const float*  __restrict__ abias,
                                              bf16_t* __restrict__ Vals) {
  __shared__ __align__(32) bf16_t Plds[4][16 * 32];   // per-wave 16x32 P tile
  const int lane = threadIdx.x & 31;
  const int wid  = threadIdx.x >> 5;
  const int l16  = lane & 15, half = lane >> 4;
  const int gw   = blockIdx.x * 4 + wid;              // 4096 waves total
  const int qt   = gw & (S_ / 16 - 1);                // 128 q tiles per head
  const int bh   = gw >> 7;                           // 0..31
  const int q0   = qt * 16;

  // Q fragments: K-dim = HD = 64 -> two 32-wide steps.
  const v16bf aQ0 = load_a(Qb + ((size_t)bh * S_ + q0) * HD_ + 0,  HD_, lane);
  const v16bf aQ1 = load_a(Qb + ((size_t)bh * S_ + q0) * HD_ + 32, HD_, lane);

  float rmax[8], rsum[8];
  v8f o[4];
  v8f zf = {};
#pragma unroll
  for (int r = 0; r < 8; ++r) { rmax[r] = -1e30f; rsum[r] = 0.0f; }
#pragma unroll
  for (int t = 0; t < 4; ++t) o[t] = zf;

  const float* bias_base = abias + (size_t)bh * S_ * S_;

  for (int kc = 0; kc < S_; kc += 32) {
    // ---- S tiles: logits for columns [kc,kc+16) and [kc+16,kc+32) ----
    v8f s0 = {}, s1 = {};
    {
      v16bf bk;
      bk = load_bt(Kmb + ((size_t)bh * S_ + kc) * HD_ + 0,  HD_, lane);
      s0 = wmma_bf16(aQ0, bk, s0);
      bk = load_bt(Kmb + ((size_t)bh * S_ + kc) * HD_ + 32, HD_, lane);
      s0 = wmma_bf16(aQ1, bk, s0);
      bk = load_bt(Kmb + ((size_t)bh * S_ + kc + 16) * HD_ + 0,  HD_, lane);
      s1 = wmma_bf16(aQ0, bk, s1);
      bk = load_bt(Kmb + ((size_t)bh * S_ + kc + 16) * HD_ + 32, HD_, lane);
      s1 = wmma_bf16(aQ1, bk, s1);
    }
    // ---- bias + scale + clamp + online softmax (per row, 16-lane reductions) ----
#pragma unroll
    for (int r = 0; r < 8; ++r) {
      const int m = r + half * 8;
      const float* bp = bias_base + (size_t)(q0 + m) * S_ + kc + l16;
      float x0 = (s0[r] + __builtin_nontemporal_load(bp))      * INV_SQRT_HD;
      float x1 = (s1[r] + __builtin_nontemporal_load(bp + 16)) * INV_SQRT_HD;
      x0 = fminf(fmaxf(x0, -CLAMP_), CLAMP_);
      x1 = fminf(fmaxf(x1, -CLAMP_), CLAMP_);
      float tm = fmaxf(x0, x1);
#pragma unroll
      for (int mk = 1; mk <= 8; mk <<= 1) tm = fmaxf(tm, __shfl_xor(tm, mk, 32));
      const float nm   = fmaxf(rmax[r], tm);
      const float corr = __expf(rmax[r] - nm);
      rmax[r] = nm;
      const float p0 = __expf(x0 - nm);
      const float p1 = __expf(x1 - nm);
      float ps = p0 + p1;
#pragma unroll
      for (int mk = 1; mk <= 8; mk <<= 1) ps += __shfl_xor(ps, mk, 32);
      rsum[r] = rsum[r] * corr + ps;
#pragma unroll
      for (int t = 0; t < 4; ++t) o[t][r] *= corr;
      Plds[wid][m * 32 + l16]      = f2bf(p0);
      Plds[wid][m * 32 + 16 + l16] = f2bf(p1);
    }
    // LDS ops from the same wave are in-order; explicit wait for safety.
    asm volatile("s_wait_dscnt 0" ::: "memory");
    // ---- P * V : A = P (16x32 from LDS), B = Vt rows (contiguous) ----
    const v16bf aP = load_a(&Plds[wid][0], 32, lane);
#pragma unroll
    for (int t = 0; t < 4; ++t) {
      v16bf bv = load_bt(Vt + ((size_t)bh * HD_ + t * 16) * S_ + kc, S_, lane);
      o[t] = wmma_bf16(aP, bv, o[t]);
    }
  }

  // ---- normalize and store vals in [B, S, E] bf16, e = h*64 + d ----
  const int b = bh / H_, h = bh % H_;
#pragma unroll
  for (int r = 0; r < 8; ++r) {
    const float inv = 1.0f / rsum[r];
    const int row = q0 + r + half * 8;
    bf16_t* op = Vals + ((size_t)b * S_ + row) * E_ + h * HD_;
#pragma unroll
    for (int t = 0; t < 4; ++t) op[t * 16 + l16] = f2bf(o[t][r] * inv);
  }
}

// Output projection: out[row, f] = sum_e Vals[row,e] * Wo[f,e] + bo[f]  (f32 out)
__global__ void __launch_bounds__(256) k_out_gemm(const bf16_t* __restrict__ Ab,
                                                  const bf16_t* __restrict__ Wob,
                                                  const float*  __restrict__ bo,
                                                  float* __restrict__ out) {
  const int lane = threadIdx.x & 31;
  const int wid  = threadIdx.x >> 5;
  const int tile = blockIdx.x * 8 + wid;     // 16384 tiles
  const int mt   = tile & 255;               // M = 4096
  const int nt   = tile >> 8;                // N = 1024 -> 64 tiles
  const int row0 = mt * 16, col0 = nt * 16;

  v8f acc = {};
#pragma unroll 4
  for (int k0 = 0; k0 < E_; k0 += 32) {
    v16bf a = load_a (Ab  + (size_t)row0 * E_ + k0, E_, lane);
    v16bf b = load_bt(Wob + (size_t)col0 * E_ + k0, E_, lane);
    acc = wmma_bf16(a, b, acc);
  }

  const int l16 = lane & 15, half = lane >> 4;
  const int f = col0 + l16;
  const float bias = bo[f];
#pragma unroll
  for (int r = 0; r < 8; ++r) {
    const int row = row0 + r + half * 8;
    out[(size_t)row * E_ + f] = acc[r] + bias;
  }
}

// ---------- launch ----------

extern "C" void kernel_launch(void* const* d_in, const int* in_sizes, int n_in,
                              void* d_out, int out_size, void* d_ws, size_t ws_size,
                              hipStream_t stream) {
  (void)in_sizes; (void)n_in; (void)out_size; (void)ws_size;
  const float* x     = (const float*)d_in[0];
  const float* abias = (const float*)d_in[1];
  const float* Wqkv  = (const float*)d_in[2];
  const float* bqkv  = (const float*)d_in[3];
  const float* Wo    = (const float*)d_in[4];
  const float* bo    = (const float*)d_in[5];
  float* out = (float*)d_out;

  // Workspace layout (bf16 staging, ~48 MB total)
  char* ws = (char*)d_ws;
  size_t off = 0;
  auto alloc = [&](size_t elems) {
    bf16_t* p = (bf16_t*)(ws + off);
    off += elems * sizeof(bf16_t);
    return p;
  };
  bf16_t* Xb   = alloc((size_t)B_ * S_ * E_);        // 8 MB
  bf16_t* Wqb  = alloc((size_t)3 * E_ * E_);         // 6 MB
  bf16_t* Wob  = alloc((size_t)E_ * E_);             // 2 MB
  bf16_t* Qb   = alloc((size_t)B_ * H_ * S_ * HD_);  // 8 MB
  bf16_t* Kmb  = alloc((size_t)B_ * H_ * S_ * HD_);  // 8 MB
  bf16_t* Vt   = alloc((size_t)B_ * H_ * S_ * HD_);  // 8 MB (transposed V)
  bf16_t* Vals = alloc((size_t)B_ * S_ * E_);        // 8 MB

  k_cvt<<<1024, 256, 0, stream>>>(x,    Xb,  B_ * S_ * E_);
  k_cvt<<<1024, 256, 0, stream>>>(Wqkv, Wqb, 3 * E_ * E_);
  k_cvt<<< 512, 256, 0, stream>>>(Wo,   Wob, E_ * E_);

  k_qkv_gemm<<<6144, 256, 0, stream>>>(Xb, Wqb, bqkv, Qb, Kmb, Vt);
  k_attn    <<<1024, 128, 0, stream>>>(Qb, Kmb, Vt, abias, Vals);
  k_out_gemm<<<2048, 256, 0, stream>>>(Vals, Wob, bo, out);
}
